// AGFAttention_44435731644439
// MI455X (gfx1250) — matrix operand order repaired and checked by reading, not compiled, not run on hardware
//
#include <hip/hip_runtime.h>
#include <cstdint>
#include <cstddef>

// ---------------- CDNA5 WMMA types ----------------
typedef __attribute__((ext_vector_type(16))) __bf16 v16bf;
typedef __attribute__((ext_vector_type(8)))  float  v8f;

#define B_    2
#define N_    2048
#define DIM_  1024
#define H_    16
#define DH_   64
#define INNER_ 1024
#define QKVN_ 3072
#define TOPK_ 64
#define SCALE_ 0.125f   // DH^-0.5

static __device__ __forceinline__ unsigned short f2bf(float f) {
  unsigned u = __float_as_uint(f);
  unsigned r = (u + 0x7FFFu + ((u >> 16) & 1u)) >> 16;   // RNE
  return (unsigned short)r;
}
static __device__ __forceinline__ unsigned f2key(float f) {
  unsigned u = __float_as_uint(f);
  return (u & 0x80000000u) ? ~u : (u | 0x80000000u);     // monotonic total order
}

union FragBF { uint4 u4[2]; v16bf v; };

// A fragment (16x32 bf16): lane = {m = lane&15, half = lane>>4}
// per-lane data = row m, k in [k0+16*half, +16)
static __device__ __forceinline__ v16bf load_a_frag(const unsigned short* A, int lda,
                                                    int m0, int k0, int lane) {
  int m = lane & 15, hf = lane >> 4;
  const unsigned short* p = A + (size_t)(m0 + m) * lda + k0 + 16 * hf;
  FragBF f;
  f.u4[0] = *reinterpret_cast<const uint4*>(p);
  f.u4[1] = *reinterpret_cast<const uint4*>(p + 8);
  return f.v;
}
// B fragment (32x16 bf16) from pre-transposed B (BT row = output column, contiguous in k):
// per-lane data = column n, k in [k0+16*half, +16)
static __device__ __forceinline__ v16bf load_b_frag(const unsigned short* BT, int ldb,
                                                    int n0, int k0, int lane) {
  int n = lane & 15, hf = lane >> 4;
  const unsigned short* p = BT + (size_t)(n0 + n) * ldb + k0 + 16 * hf;
  FragBF f;
  f.u4[0] = *reinterpret_cast<const uint4*>(p);
  f.u4[1] = *reinterpret_cast<const uint4*>(p + 8);
  return f.v;
}
static __device__ __forceinline__ v8f wmma_bf16(v16bf a, v16bf b, v8f c) {
  return __builtin_amdgcn_wmma_f32_16x16x32_bf16(false, a, false, b, (short)0, c,
                                                 false, false);
}

static __device__ __forceinline__ unsigned waveSumU(unsigned v) {
  for (int off = 16; off; off >>= 1) v += __shfl_xor(v, off, 32);
  return v;
}
static __device__ __forceinline__ float waveSumF(float v) {
  for (int off = 16; off; off >>= 1) v += __shfl_xor(v, off, 32);
  return v;
}
static __device__ __forceinline__ float waveMaxF(float v) {
  for (int off = 16; off; off >>= 1) v = fmaxf(v, __shfl_xor(v, off, 32));
  return v;
}

// ---------------- conversion kernels ----------------
__global__ void cvt_bf16_k(const float* __restrict__ src, unsigned short* __restrict__ dst, int n) {
  for (int i = blockIdx.x * blockDim.x + threadIdx.x; i < n; i += gridDim.x * blockDim.x)
    dst[i] = f2bf(src[i]);
}
// src [R,C] fp32 row-major -> dst [C,R] bf16 row-major (transpose)
__global__ void cvtT_bf16_k(const float* __restrict__ src, unsigned short* __restrict__ dst,
                            int R, int C) {
  int n = R * C;
  for (int i = blockIdx.x * blockDim.x + threadIdx.x; i < n; i += gridDim.x * blockDim.x) {
    int r = i / C, c = i - r * C;
    dst[(size_t)c * R + r] = f2bf(src[i]);
  }
}

// ---------------- QKV projection GEMM ----------------
// qkv[4096,3072] = x_bf[4096,1024] @ WqkvT_bf[3072,1024]^T + bqkv
// epilogue scatters to q_bf,k_bf (head-major bf16) and v_f (fp32).
__global__ void qkv_gemm_k(const unsigned short* __restrict__ xbf,
                           const unsigned short* __restrict__ wT,
                           const float* __restrict__ bqkv,
                           unsigned short* __restrict__ qbf,
                           unsigned short* __restrict__ kbf,
                           float* __restrict__ vf) {
  int wave = threadIdx.x >> 5, lane = threadIdx.x & 31;
  int m0 = blockIdx.x * 16;
  int n0 = (blockIdx.y * 4 + wave) * 16;
  v8f acc = {0.f, 0.f, 0.f, 0.f, 0.f, 0.f, 0.f, 0.f};
  for (int k0 = 0; k0 < DIM_; k0 += 32) {
    v16bf a = load_a_frag(xbf, DIM_, m0, k0, lane);
    v16bf b = load_b_frag(wT, DIM_, n0, k0, lane);
    acc = wmma_bf16(a, b, acc);
  }
  int col = n0 + (lane & 15);
  int hf = lane >> 4;
  float bias = bqkv[col];
  int sel = col >> 10, jj = col & 1023, hh = jj >> 6, dd = jj & 63;
#pragma unroll
  for (int r = 0; r < 8; ++r) {
    int row = m0 + r + 8 * hf;               // 0..4095 = b*2048+n
    int b_ = row >> 11, n_ = row & 2047;
    float val = acc[r] + bias;
    size_t dst = (((size_t)(b_ * H_ + hh)) * N_ + n_) * DH_ + dd;
    if (sel == 0)       qbf[dst] = f2bf(val);
    else if (sel == 1)  kbf[dst] = f2bf(val);
    else                vf[dst]  = val;
  }
}

// ---------------- scores + exact top-64 + softmax -> sparse (idx,w) ----------------
// grid: (N/16, B*H); 256 threads (8 waves); dynamic LDS: 16*2048 fp32 = 128 KB
__global__ void scores_topk_k(const unsigned short* __restrict__ qbf,
                              const unsigned short* __restrict__ kbf,
                              int* __restrict__ tidx, float* __restrict__ tw) {
  extern __shared__ float s[];               // [16][2048]
  int wave = threadIdx.x >> 5, lane = threadIdx.x & 31;
  int qt = blockIdx.x, bh = blockIdx.y;
  const unsigned short* qbase = qbf + (size_t)bh * N_ * DH_;
  const unsigned short* kbase = kbf + (size_t)bh * N_ * DH_;
  int m0 = qt * 16;

  v16bf a0 = load_a_frag(qbase, DH_, m0, 0, lane);
  v16bf a1 = load_a_frag(qbase, DH_, m0, 32, lane);
  int hf = lane >> 4, nn = lane & 15;
  for (int kt = wave; kt < N_ / 16; kt += 8) {
    int n0 = kt * 16;
    v16bf b0 = load_b_frag(kbase, DH_, n0, 0, lane);
    v16bf b1 = load_b_frag(kbase, DH_, n0, 32, lane);
    v8f acc = {0.f, 0.f, 0.f, 0.f, 0.f, 0.f, 0.f, 0.f};
    acc = wmma_bf16(a0, b0, acc);
    acc = wmma_bf16(a1, b1, acc);
    int col = n0 + nn;
#pragma unroll
    for (int r = 0; r < 8; ++r)
      s[(size_t)(r + 8 * hf) * N_ + col] = acc[r] * SCALE_;
  }
  __syncthreads();

  // one wave handles 2 rows; exact kth-largest via bisection on monotonic uint keys
  for (int rr = 0; rr < 2; ++rr) {
    int row = wave * 2 + rr;
    const float* srow = s + (size_t)row * N_;
    unsigned lo = 0u, hi = 0xFFFFFFFFu;
    while (lo < hi) {
      unsigned d = hi - lo;
      unsigned mid = lo + (d >> 1) + (d & 1u);            // ceil midpoint, overflow-safe
      unsigned c = 0;
      for (int j = 0; j < N_ / 32; ++j)
        c += (f2key(srow[j * 32 + lane]) >= mid) ? 1u : 0u;
      c = waveSumU(c);
      if (c >= TOPK_) lo = mid; else hi = mid - 1;
    }
    unsigned T = lo;                                      // key of 64th largest
    float m = -3.0e38f;
    for (int j = 0; j < N_ / 32; ++j) m = fmaxf(m, srow[j * 32 + lane]);
    m = waveMaxF(m);
    float dsum = 0.f;
    for (int j = 0; j < N_ / 32; ++j) {
      float f = srow[j * 32 + lane];
      if (f2key(f) >= T) dsum += __expf(f - m);
    }
    dsum = waveSumF(dsum);
    float inv = 1.0f / dsum;

    // deterministic compaction: lane owns contiguous cols [lane*64, lane*64+64)
    int kept = 0;
    for (int j = 0; j < 64; ++j)
      kept += (f2key(srow[lane * 64 + j]) >= T) ? 1 : 0;
    unsigned incl = (unsigned)kept;
    for (int off = 1; off < 32; off <<= 1) {
      unsigned nv = __shfl_up(incl, off, 32);
      if (lane >= off) incl += nv;
    }
    int pos = (int)(incl - (unsigned)kept);
    size_t grow = (size_t)bh * N_ + m0 + row;
    int* ti = tidx + grow * TOPK_;
    float* twp = tw + grow * TOPK_;
    for (int j = 0; j < 64; ++j) {
      int c = lane * 64 + j;
      float f = srow[c];
      if (f2key(f) >= T) {
        if (pos < TOPK_) { ti[pos] = c; twp[pos] = __expf(f - m) * inv; }
        ++pos;
      }
    }
  }
}

// ---------------- sparse hop: v_next = attn @ v_cur ; res (+)= alpha_i * v_next ----
__global__ void spmv_hop_k(const float* __restrict__ vin, float* __restrict__ vout,
                           float* __restrict__ res,
                           const int* __restrict__ tidx, const float* __restrict__ tw,
                           const float* __restrict__ alphas_raw, int ord, int first) {
  int wave = threadIdx.x >> 5, lane = threadIdx.x & 31;
  size_t row = (size_t)blockIdx.x * 8 + wave;            // 0..B*H*N-1 = bh*2048+n
  int h = (int)((row >> 11) & 15);
  float ar = alphas_raw[ord * H_ + h];
  float alpha = 1.0f / (1.0f + __expf(-ar));
  const int* ti = tidx + row * TOPK_;
  const float* twp = tw + row * TOPK_;
  const float* vbase = vin + ((row >> 11) << 11) * DH_;  // start of this (b,h) block
  float a0 = 0.f, a1 = 0.f;
#pragma unroll 4
  for (int j = 0; j < TOPK_; ++j) {
    int idx = ti[j];
    float w = twp[j];
    const float* vp = vbase + (size_t)idx * DH_;
    a0 += w * vp[lane];
    a1 += w * vp[lane + 32];
  }
  size_t o = row * DH_;
  vout[o + lane] = a0;
  vout[o + lane + 32] = a1;
  if (first) { res[o + lane] = alpha * a0;  res[o + lane + 32] = alpha * a1; }
  else       { res[o + lane] += alpha * a0; res[o + lane + 32] += alpha * a1; }
}

// ---------------- res [B,H,N,DH] fp32 -> res_bf [B,N,H*DH] bf16 ----------------
__global__ void res_cvt_k(const float* __restrict__ res, unsigned short* __restrict__ rbf) {
  int n = B_ * H_ * N_ * DH_;
  for (int i = blockIdx.x * blockDim.x + threadIdx.x; i < n; i += gridDim.x * blockDim.x) {
    int bh = i >> 17;                // /(N_*DH_)
    int b = bh >> 4, h = bh & 15;
    int nn = (i >> 6) & (N_ - 1);
    int d = i & 63;
    rbf[((size_t)(b * N_ + nn)) * INNER_ + h * DH_ + d] = f2bf(res[i]);
  }
}

// ---------------- output projection GEMM ----------------
__global__ void out_gemm_k(const unsigned short* __restrict__ rbf,
                           const unsigned short* __restrict__ woutT,
                           const float* __restrict__ bout,
                           float* __restrict__ out) {
  int wave = threadIdx.x >> 5, lane = threadIdx.x & 31;
  int m0 = blockIdx.x * 16;
  int n0 = (blockIdx.y * 4 + wave) * 16;
  v8f acc = {0.f, 0.f, 0.f, 0.f, 0.f, 0.f, 0.f, 0.f};
  for (int k0 = 0; k0 < INNER_; k0 += 32) {
    v16bf a = load_a_frag(rbf, INNER_, m0, k0, lane);
    v16bf b = load_b_frag(woutT, INNER_, n0, k0, lane);
    acc = wmma_bf16(a, b, acc);
  }
  int col = n0 + (lane & 15);
  int hf = lane >> 4;
  float bias = bout[col];
#pragma unroll
  for (int r = 0; r < 8; ++r) {
    int row = m0 + r + 8 * hf;
    out[(size_t)row * DIM_ + col] = acc[r] + bias;
  }
}

// ---------------- host launcher ----------------
extern "C" void kernel_launch(void* const* d_in, const int* in_sizes, int n_in,
                              void* d_out, int out_size, void* d_ws, size_t ws_size,
                              hipStream_t stream) {
  const float* x          = (const float*)d_in[0];   // [B,N,DIM]
  const float* Wqkv       = (const float*)d_in[1];   // [DIM,3*INNER]
  const float* bqkv       = (const float*)d_in[2];   // [3*INNER]
  const float* Wout       = (const float*)d_in[3];   // [INNER,DIM]
  const float* bout       = (const float*)d_in[4];   // [DIM]
  const float* alphas_raw = (const float*)d_in[5];   // [4,H]
  float* out = (float*)d_out;                        // [B,N,DIM] fp32

  // workspace carve-up (256B aligned)
  char* p = (char*)d_ws;
  auto take = [&](size_t bytes) { char* r = p; p += (bytes + 255) & ~(size_t)255; return r; };
  const size_t ROWS = (size_t)B_ * H_ * N_;          // 65536
  unsigned short* x_bf   = (unsigned short*)take((size_t)B_ * N_ * DIM_ * 2);
  unsigned short* wqkvT  = (unsigned short*)take((size_t)QKVN_ * DIM_ * 2);
  unsigned short* woutT  = (unsigned short*)take((size_t)DIM_ * INNER_ * 2);
  unsigned short* q_bf   = (unsigned short*)take(ROWS * DH_ * 2);
  unsigned short* k_bf   = (unsigned short*)take(ROWS * DH_ * 2);
  float*          v_f    = (float*)take(ROWS * DH_ * 4);
  int*            tidx   = (int*)take(ROWS * TOPK_ * 4);
  float*          tw     = (float*)take(ROWS * TOPK_ * 4);
  float*          v1     = (float*)take(ROWS * DH_ * 4);
  float*          v2     = (float*)take(ROWS * DH_ * 4);
  float*          res    = (float*)take(ROWS * DH_ * 4);
  unsigned short* res_bf = (unsigned short*)take((size_t)B_ * N_ * INNER_ * 2);

  // 1) precision conversions (+ weight transposes for contiguous B-fragments)
  cvt_bf16_k <<<4096, 256, 0, stream>>>(x, x_bf, B_ * N_ * DIM_);
  cvtT_bf16_k<<<4096, 256, 0, stream>>>(Wqkv, wqkvT, DIM_, QKVN_);
  cvtT_bf16_k<<<2048, 256, 0, stream>>>(Wout, woutT, INNER_, DIM_);

  // 2) QKV projection (bf16 WMMA, fp32 accumulate)
  qkv_gemm_k<<<dim3(B_ * N_ / 16, QKVN_ / 64), 128, 0, stream>>>(
      x_bf, wqkvT, bqkv, q_bf, k_bf, v_f);

  // 3) scores strip (WMMA into 128KB LDS) + exact top-64 + softmax -> sparse attn
  scores_topk_k<<<dim3(N_ / 16, B_ * H_), 256, 16 * N_ * sizeof(float), stream>>>(
      q_bf, k_bf, tidx, tw);

  // 4) polynomial graph filter: three 64-nnz sparse hops
  int blocks = (int)(ROWS / 8);
  spmv_hop_k<<<blocks, 256, 0, stream>>>(v_f, v1, res, tidx, tw, alphas_raw, 1, 1);
  spmv_hop_k<<<blocks, 256, 0, stream>>>(v1, v2, res, tidx, tw, alphas_raw, 2, 0);
  spmv_hop_k<<<blocks, 256, 0, stream>>>(v2, v1, res, tidx, tw, alphas_raw, 3, 0);

  // 5) output projection
  res_cvt_k<<<4096, 256, 0, stream>>>(res, res_bf);
  out_gemm_k<<<dim3(B_ * N_ / 16, DIM_ / 64), 128, 0, stream>>>(
      res_bf, woutT, bout, out);
}
